// SelfAttentionBlock_51754355917105
// MI455X (gfx1250) — matrix operand order repaired and checked by reading, not compiled, not run on hardware
//
#include <hip/hip_runtime.h>
#include <hip/hip_bf16.h>

typedef unsigned int u32x4 __attribute__((ext_vector_type(4)));
typedef float        v8f   __attribute__((ext_vector_type(8)));
typedef __bf16       v16bf __attribute__((ext_vector_type(16)));

union Frag { u32x4 q[2]; v16bf v; };

#define WMMA_BF16(A, B, C) \
  __builtin_amdgcn_wmma_f32_16x16x32_bf16(false, (A), false, (B), (short)0, (C), false, false)

static __device__ __forceinline__ float red16_max(float t) {
  t = fmaxf(t, __shfl_xor(t, 1, 32));
  t = fmaxf(t, __shfl_xor(t, 2, 32));
  t = fmaxf(t, __shfl_xor(t, 4, 32));
  t = fmaxf(t, __shfl_xor(t, 8, 32));
  return t;
}
static __device__ __forceinline__ float red16_sum(float t) {
  t += __shfl_xor(t, 1, 32);
  t += __shfl_xor(t, 2, 32);
  t += __shfl_xor(t, 4, 32);
  t += __shfl_xor(t, 8, 32);
  return t;
}

// ---------------------------------------------------------------------------
// Kernel 1: spectral norm power iteration + bf16 weight conversion. 1 block.
// ---------------------------------------------------------------------------
__global__ __launch_bounds__(256) void prep_kernel(
    const float* __restrict__ Wk, const float* __restrict__ u,
    const float* __restrict__ Wv, const float* __restrict__ Ww,
    __bf16* __restrict__ Wqb, __bf16* __restrict__ Wvb, __bf16* __restrict__ Wwb) {
  __shared__ float sv[256];
  __shared__ float red[256];
  int tid = threadIdx.x;

  // v = Wk^T u  (256 elems)
  float acc = 0.f;
  for (int k = 0; k < 128; ++k) acc += Wk[k * 256 + tid] * u[k];
  sv[tid] = acc;
  red[tid] = acc * acc;
  __syncthreads();
  #pragma unroll
  for (int s = 128; s > 0; s >>= 1) {
    if (tid < s) red[tid] += red[tid + s];
    __syncthreads();
  }
  float vn = sqrtf(red[0]) + 1e-12f;
  __syncthreads();
  sv[tid] = sv[tid] / vn;
  __syncthreads();

  // u2raw = Wk v (128 elems)
  float u2 = 0.f;
  if (tid < 128) {
    for (int c = 0; c < 256; ++c) u2 += Wk[tid * 256 + c] * sv[c];
  }
  red[tid] = (tid < 128) ? u2 * u2 : 0.f;
  __syncthreads();
  #pragma unroll
  for (int s = 128; s > 0; s >>= 1) {
    if (tid < s) red[tid] += red[tid + s];
    __syncthreads();
  }
  float n2 = red[0];
  float sigma = n2 / (sqrtf(n2) + 1e-12f);   // u2_hat . (Wk v)
  float inv = 1.f / sigma;

  // bf16 copies: Wq = Wk/sigma (128x256), Wv (128x256), Ww (256x128)
  for (int i = tid; i < 128 * 256; i += 256) Wqb[i] = (__bf16)(Wk[i] * inv);
  for (int i = tid; i < 128 * 256; i += 256) Wvb[i] = (__bf16)Wv[i];
  for (int i = tid; i < 256 * 128; i += 256) Wwb[i] = (__bf16)Ww[i];
}

// ---------------------------------------------------------------------------
// Kernel 2: maxpool 2x2x2 -> bf16, position-major [B, N=4096, C=256]
// ---------------------------------------------------------------------------
__global__ __launch_bounds__(256) void pool_kernel(const float* __restrict__ x,
                                                   __bf16* __restrict__ Xf) {
  int idx = blockIdx.x * 256 + threadIdx.x;        // 4*256*4096 = 2^22 threads
  int b = idx >> 20;
  int rem = idx & ((1 << 20) - 1);
  int c = rem >> 12;
  int n = rem & 4095;
  int z = n >> 8, y = (n >> 4) & 15, xc = n & 15;
  const float* p = x + (size_t)(b * 256 + c) * 32768 + (size_t)(2 * z) * 1024
                     + (2 * y) * 32 + 2 * xc;
  float m = -3.0e38f;
  #pragma unroll
  for (int dz = 0; dz < 2; ++dz)
    #pragma unroll
    for (int dy = 0; dy < 2; ++dy) {
      const float* q = p + dz * 1024 + dy * 32;
      m = fmaxf(m, fmaxf(q[0], q[1]));
    }
  Xf[((size_t)b * 4096 + n) * 256 + c] = (__bf16)m;
}

// ---------------------------------------------------------------------------
// Kernel 3: projections.  Q[b,n,128] = (Wq Xf + bk)*kc^-0.25 (bf16)
//                         Vt[b,128,n] = Wv Xf + bv           (bf16, transposed)
// One wave = 16 rows x 128 cols, K=256. 8 waves/block, 128 blocks.
// ---------------------------------------------------------------------------
__global__ __launch_bounds__(256) void proj_kernel(
    const __bf16* __restrict__ Xf, const __bf16* __restrict__ Wqb,
    const __bf16* __restrict__ Wvb, const float* __restrict__ bk,
    const float* __restrict__ bv, __bf16* __restrict__ Q, __bf16* __restrict__ Vt) {
  int lane = threadIdx.x & 31, wave = threadIdx.x >> 5;
  int b = blockIdx.x >> 5;
  int n0 = (blockIdx.x & 31) * 128 + wave * 16;
  int row = lane & 15, hi = lane >> 4, ao = hi * 8, bo = hi * 16;
  const __bf16* Xb = Xf + (size_t)b * 4096 * 256;

  v8f aq[8], av[8];
  #pragma unroll
  for (int t = 0; t < 8; ++t) { aq[t] = (v8f){0,0,0,0,0,0,0,0}; av[t] = (v8f){0,0,0,0,0,0,0,0}; }

  #pragma unroll
  for (int ck = 0; ck < 8; ++ck) {
    Frag a;
    const __bf16* xb = Xb + (size_t)(n0 + row) * 256 + ck * 32;
    a.q[0] = *(const u32x4*)(xb + bo);
    a.q[1] = *(const u32x4*)(xb + bo + 8);
    #pragma unroll
    for (int t = 0; t < 8; ++t) {
      Frag bq, bvv;
      const __bf16* wq = Wqb + (size_t)(t * 16 + row) * 256 + ck * 32 + bo;
      bq.q[0] = *(const u32x4*)wq;  bq.q[1] = *(const u32x4*)(wq + 8);
      const __bf16* wv = Wvb + (size_t)(t * 16 + row) * 256 + ck * 32 + bo;
      bvv.q[0] = *(const u32x4*)wv; bvv.q[1] = *(const u32x4*)(wv + 8);
      aq[t] = WMMA_BF16(a.v, bq.v, aq[t]);
      av[t] = WMMA_BF16(a.v, bvv.v, av[t]);
    }
  }

  const float qscale = 0.29730177875068026f;  // 128^-0.25 ; (s*q).(s*k)=sim/sqrt(128)
  #pragma unroll
  for (int t = 0; t < 8; ++t) {
    int kc = t * 16 + row;
    float bkv = bk[kc], bvvv = bv[kc];
    #pragma unroll
    for (int r = 0; r < 8; ++r) {
      int n = n0 + r + ao;                       // C-tile row M = r + 8*hi
      Q[((size_t)b * 4096 + n) * 128 + kc]  = (__bf16)((aq[t][r] + bkv) * qscale);
      Vt[((size_t)b * 128 + kc) * 4096 + n] = (__bf16)(av[t][r] + bvvv);
    }
  }
}

// ---------------------------------------------------------------------------
// Kernel 4: flash attention.  One wave = 16 query rows; online softmax over
// 32-key chunks; ctx written bf16 [b, n, 128].  4 waves/block, 256 blocks.
// ---------------------------------------------------------------------------
__global__ __launch_bounds__(128) void attn_kernel(
    const __bf16* __restrict__ Q, const __bf16* __restrict__ Vt,
    __bf16* __restrict__ Ctx) {
  __shared__ __bf16 pbuf[4][16][32];             // per-wave P staging tile
  int lane = threadIdx.x & 31, wave = threadIdx.x >> 5;
  int tile = blockIdx.x * 4 + wave;              // 1024 tiles
  int b = tile >> 8;
  int n0 = (tile & 255) << 4;
  int row = lane & 15, hi = lane >> 4, ao = hi * 8, bo = hi * 16;
  const __bf16* Qb = Q + (size_t)b * 4096 * 128;
  const __bf16* Vb = Vt + (size_t)b * 128 * 4096;

  // Q A-tiles: 16 rows x 128 features (4 chunks of K=32)
  Frag qa[4];
  #pragma unroll
  for (int ck = 0; ck < 4; ++ck) {
    const __bf16* base = Qb + (size_t)(n0 + row) * 128 + ck * 32;
    qa[ck].q[0] = *(const u32x4*)(base + bo);
    qa[ck].q[1] = *(const u32x4*)(base + bo + 8);
  }

  v8f acc[8];
  #pragma unroll
  for (int t = 0; t < 8; ++t) acc[t] = (v8f){0,0,0,0,0,0,0,0};
  float mrun[8], lrun[8];
  #pragma unroll
  for (int r = 0; r < 8; ++r) { mrun[r] = -3.0e38f; lrun[r] = 0.f; }

  for (int k0 = 0; k0 < 4096; k0 += 32) {
    // scores: S = Q K^T for 32 keys (two 16-col WMMA tiles, K=128)
    v8f clo = (v8f){0,0,0,0,0,0,0,0}, chi = (v8f){0,0,0,0,0,0,0,0};
    #pragma unroll
    for (int ck = 0; ck < 4; ++ck) {
      Frag bl, bh;
      const __bf16* kl = Qb + (size_t)(k0 + row) * 128 + ck * 32 + bo;
      bl.q[0] = *(const u32x4*)kl; bl.q[1] = *(const u32x4*)(kl + 8);
      const __bf16* kh = Qb + (size_t)(k0 + 16 + row) * 128 + ck * 32 + bo;
      bh.q[0] = *(const u32x4*)kh; bh.q[1] = *(const u32x4*)(kh + 8);
      clo = WMMA_BF16(qa[ck].v, bl.v, clo);
      chi = WMMA_BF16(qa[ck].v, bh.v, chi);
    }

    // online softmax update (rows live in 16-lane halves, col = lane%16)
    float scale[8];
    #pragma unroll
    for (int r = 0; r < 8; ++r) {
      float t = red16_max(fmaxf(clo[r], chi[r]));
      float mn = fmaxf(mrun[r], t);
      scale[r] = __expf(mrun[r] - mn);
      mrun[r] = mn;
      float plo = __expf(clo[r] - mn);
      float phi = __expf(chi[r] - mn);
      pbuf[wave][r + ao][row]      = (__bf16)plo;
      pbuf[wave][r + ao][row + 16] = (__bf16)phi;
      lrun[r] = lrun[r] * scale[r] + red16_sum(plo + phi);
    }
    #pragma unroll
    for (int t = 0; t < 8; ++t)
      #pragma unroll
      for (int r = 0; r < 8; ++r) acc[t][r] *= scale[r];

    // C-layout -> A-layout bounce through LDS (same-wave, DS kept in order)
    asm volatile("s_wait_dscnt 0x0" ::: "memory");
    Frag pa;
    pa.q[0] = *(const u32x4*)&pbuf[wave][row][bo];
    pa.q[1] = *(const u32x4*)&pbuf[wave][row][bo + 8];

    // ctx += P @ V (K = 32 keys; Vt gives contiguous K per lane)
    #pragma unroll
    for (int t = 0; t < 8; ++t) {
      Frag vb;
      const __bf16* vp = Vb + (size_t)(t * 16 + row) * 4096 + k0 + bo;
      vb.q[0] = *(const u32x4*)vp; vb.q[1] = *(const u32x4*)(vp + 8);
      acc[t] = WMMA_BF16(pa.v, vb.v, acc[t]);
    }
  }

  __bf16* Cb = Ctx + (size_t)b * 4096 * 128;
  #pragma unroll
  for (int t = 0; t < 8; ++t) {
    #pragma unroll
    for (int r = 0; r < 8; ++r) {
      int n = n0 + r + ao;
      Cb[(size_t)n * 128 + t * 16 + row] = (__bf16)(acc[t][r] / lrun[r]);
    }
  }
}

// ---------------------------------------------------------------------------
// Kernel 5: out = Ww @ ctx^T + bw  ->  fp32 [B, 256, 4096]
// One wave = 16 oc x 128 positions, K=128. 8 waves/block, 256 blocks.
// ---------------------------------------------------------------------------
__global__ __launch_bounds__(256) void outproj_kernel(
    const __bf16* __restrict__ Wwb, const __bf16* __restrict__ Ctx,
    const float* __restrict__ bw, float* __restrict__ out) {
  int lane = threadIdx.x & 31, wave = threadIdx.x >> 5;
  int job = blockIdx.x * 8 + wave;               // 2048 jobs
  int b = job >> 9;
  int rem = job & 511;
  int m0 = (rem >> 5) << 4;                      // oc tile
  int n0 = (rem & 31) << 7;                      // position block
  int row = lane & 15, hi = lane >> 4, ao = hi * 8, bo = hi * 16;

  v8f acc[8];
  #pragma unroll
  for (int t = 0; t < 8; ++t) acc[t] = (v8f){0,0,0,0,0,0,0,0};

  #pragma unroll
  for (int ck = 0; ck < 4; ++ck) {
    Frag wa;
    const __bf16* wp = Wwb + (size_t)(m0 + row) * 128 + ck * 32;
    wa.q[0] = *(const u32x4*)(wp + bo);
    wa.q[1] = *(const u32x4*)(wp + bo + 8);
    #pragma unroll
    for (int t = 0; t < 8; ++t) {
      Frag cb;
      const __bf16* cp = Ctx + ((size_t)b * 4096 + n0 + t * 16 + row) * 128 + ck * 32 + bo;
      cb.q[0] = *(const u32x4*)cp; cb.q[1] = *(const u32x4*)(cp + 8);
      acc[t] = WMMA_BF16(wa.v, cb.v, acc[t]);
    }
  }

  #pragma unroll
  for (int t = 0; t < 8; ++t) {
    #pragma unroll
    for (int r = 0; r < 8; ++r) {
      int oc = m0 + r + ao;
      int n = n0 + t * 16 + row;
      out[((size_t)b * 256 + oc) * 4096 + n] = acc[t][r] + bw[oc];
    }
  }
}

// ---------------------------------------------------------------------------
extern "C" void kernel_launch(void* const* d_in, const int* in_sizes, int n_in,
                              void* d_out, int out_size, void* d_ws, size_t ws_size,
                              hipStream_t stream) {
  const float* x  = (const float*)d_in[0];
  const float* Wk = (const float*)d_in[1];
  const float* bk = (const float*)d_in[2];
  const float* Wv = (const float*)d_in[3];
  const float* bv = (const float*)d_in[4];
  const float* Ww = (const float*)d_in[5];
  const float* bw = (const float*)d_in[6];
  const float* u  = (const float*)d_in[7];
  float* out = (float*)d_out;

  char* ws = (char*)d_ws;
  __bf16* Wqb = (__bf16*)(ws);                       //  64 KB  (128x256)
  __bf16* Wvb = (__bf16*)(ws + (64 << 10));          //  64 KB  (128x256)
  __bf16* Wwb = (__bf16*)(ws + (128 << 10));         //  64 KB  (256x128)
  __bf16* Xf  = (__bf16*)(ws + (192 << 10));         // 8 MB    [4,4096,256]
  char* p = ws + (192 << 10) + (size_t)4 * 4096 * 256 * 2;
  __bf16* Qm  = (__bf16*)(p);                        // 4 MB    [4,4096,128]
  __bf16* Vt  = (__bf16*)(p + (size_t)4 * 4096 * 128 * 2);
  __bf16* Ctx = (__bf16*)(p + (size_t)8 * 4096 * 128 * 2);

  prep_kernel<<<1, 256, 0, stream>>>(Wk, u, Wv, Ww, Wqb, Wvb, Wwb);
  pool_kernel<<<(4 * 256 * 4096) / 256, 256, 0, stream>>>(x, Xf);
  proj_kernel<<<128, 256, 0, stream>>>(Xf, Wqb, Wvb, bk, bv, Qm, Vt);
  attn_kernel<<<256, 128, 0, stream>>>(Qm, Vt, Ctx);
  outproj_kernel<<<256, 256, 0, stream>>>(Wwb, Ctx, bw, out);
}
